// MSA_11416023073086
// MI455X (gfx1250) — matrix operand order repaired and checked by reading, not compiled, not run on hardware
//
#include <hip/hip_runtime.h>
#include <hip/hip_bf16.h>

typedef _Float16 half_t;
typedef __attribute__((ext_vector_type(16))) _Float16 v16h;
typedef __attribute__((ext_vector_type(8)))  _Float16 v8h;
typedef __attribute__((ext_vector_type(8)))  float    v8f;

#define NN   4096
#define DIMM 768
#define HH   12
#define HDD  64

#if defined(__has_builtin)
# if __has_builtin(__builtin_amdgcn_global_load_async_to_lds_b128) && \
     __has_builtin(__builtin_amdgcn_s_wait_asynccnt)
#  define USE_ASYNC_LDS 1
# else
#  define USE_ASYNC_LDS 0
# endif
#else
# define USE_ASYNC_LDS 0
#endif

// Builtin signature (from hipcc diagnostic): param0 is
// 'int __attribute__((vector_size(16))) __device__ *' -> v4i in global AS,
// param1 the LDS-side v4i pointer, then imm offset / imm cpol.
typedef int v4i __attribute__((vector_size(16)));
typedef __attribute__((address_space(1))) v4i gv4i_t;
typedef __attribute__((address_space(3))) v4i lv4i_t;

// Copy 16 bytes global -> LDS. Async DMA path on CDNA5 (ASYNCcnt), sync fallback.
__device__ __forceinline__ void tile_copy16(const half_t* g, half_t* l) {
#if USE_ASYNC_LDS
    __builtin_amdgcn_global_load_async_to_lds_b128(
        (gv4i_t*)(unsigned long long)g,
        (lv4i_t*)(unsigned)(unsigned long long)l, 0, 0);
#else
    *(v8h*)l = *(const v8h*)g;
#endif
}
__device__ __forceinline__ void tile_wait() {
#if USE_ASYNC_LDS
    __builtin_amdgcn_s_wait_asynccnt(0);
#endif
}

__device__ __forceinline__ v8f wmma_f16(v16h a, v16h b, v8f c) {
    // D = A(16x32 f16) * B(32x16 f16) + C(16x16 f32)
    return __builtin_amdgcn_wmma_f32_16x16x32_f16(
        false, a, false, b, (short)0, c, false, false);
}

// ---------------------------------------------------------------------------
// Kernel 1: per-head QKV projection  y[h,n,e] = sum_d s[n,h*64+d]*W[h,d,e]+b[h,e]
// Output Q/K/V as f16 in workspace, laid out [H][N][64].
// ---------------------------------------------------------------------------
__global__ __launch_bounds__(256) void qkv_proj_kernel(
    const float* __restrict__ seq,
    const float* __restrict__ Wq, const float* __restrict__ bq,
    const float* __restrict__ Wk, const float* __restrict__ bk,
    const float* __restrict__ Wv, const float* __restrict__ bv,
    half_t* __restrict__ Qh, half_t* __restrict__ Kh, half_t* __restrict__ Vh)
{
    __shared__ half_t Wt[3][64][64];   // W transposed: Wt[m][e][d], f16 (24 KB)

    const int h     = blockIdx.y;
    const int ntile = blockIdx.x;
    const int tid   = threadIdx.x;
    const int lane  = tid & 31;
    const int wave  = tid >> 5;
    const int mrow  = lane & 15;       // M index within fragment
    const int hs    = lane >> 4;       // half-wave select
    const int kbase = hs * 16;

    const float* Ws[3]   = {Wq, Wk, Wv};
    const float* Bs[3]   = {bq, bk, bv};
    half_t*      Outs[3] = {Qh, Kh, Vh};

    // Cooperative transpose-load of all three 64x64 weight matrices (f32->f16)
    for (int m = 0; m < 3; ++m) {
        const float* W = Ws[m] + (size_t)h * 64 * 64;   // [d][e]
        for (int idx = tid; idx < 64 * 64; idx += 256) {
            int d = idx >> 6, e = idx & 63;
            Wt[m][e][d] = (half_t)W[idx];
        }
    }
    __syncthreads();

    const int n0 = ntile * 128 + wave * 16;

    // A fragments from seq rows (f32 -> f16), K-dim = input feature d
    v16h a0, a1;
    {
        const float* srow = seq + (size_t)(n0 + mrow) * DIMM + h * HDD;
        #pragma unroll
        for (int j = 0; j < 8; ++j) {
            a0[j]     = (half_t)srow[kbase + j];
            a0[8 + j] = (half_t)srow[kbase + 8 + j];
            a1[j]     = (half_t)srow[32 + kbase + j];
            a1[8 + j] = (half_t)srow[32 + kbase + 8 + j];
        }
    }

    for (int m = 0; m < 3; ++m) {
        half_t*      O    = Outs[m] + (size_t)h * NN * HDD;
        const float* bias = Bs[m] + h * HDD;
        #pragma unroll
        for (int t = 0; t < 4; ++t) {
            const int e = t * 16 + mrow;
            v8f c = {};
            v16h b0 = *(const v16h*)&Wt[m][e][hs * 16];
            v16h b1 = *(const v16h*)&Wt[m][e][32 + hs * 16];
            c = wmma_f16(a0, b0, c);
            c = wmma_f16(a1, b1, c);
            const float bb = bias[e];
            #pragma unroll
            for (int r = 0; r < 8; ++r) {
                const int row = n0 + hs * 8 + r;     // C layout: VGPR r <-> M=r / 8+r
                O[(size_t)row * HDD + e] = (half_t)(c[r] + bb);
            }
        }
    }
}

// ---------------------------------------------------------------------------
// Kernel 2: flash attention, double-buffered async K/V tiles in LDS.
// One wave = 16 query rows; block = 128 rows, one head; 32-key tiles.
// One barrier per key tile: wait(own ASYNCcnt) -> barrier -> issue next tile.
// ---------------------------------------------------------------------------
__global__ __launch_bounds__(256) void attn_fwd_kernel(
    const half_t* __restrict__ Qh, const half_t* __restrict__ Kh,
    const half_t* __restrict__ Vh, float* __restrict__ out)
{
    __shared__ half_t Ktile[2][32][64];  // [buf][key][d]   8 KB
    __shared__ half_t Vtile[2][32][64];  // [buf][key][d]   8 KB
    __shared__ half_t Pbuf[8][16][32];   // per-wave P tile 8 KB

    const int h     = blockIdx.y;
    const int qtile = blockIdx.x;
    const int tid   = threadIdx.x;
    const int lane  = tid & 31;
    const int wave  = tid >> 5;
    const int mrow  = lane & 15;
    const int hs    = lane >> 4;
    const int kbase = hs * 16;
    const int q0    = qtile * 128 + wave * 16;

    const half_t* Qb = Qh + (size_t)h * NN * HDD;
    const half_t* Kb = Kh + (size_t)h * NN * HDD;
    const half_t* Vb = Vh + (size_t)h * NN * HDD;

    // Q A-fragments (persist across the whole key loop)
    v16h aq0, aq1;
    {
        const half_t* qrow = Qb + (size_t)(q0 + mrow) * HDD;
        #pragma unroll
        for (int j = 0; j < 8; ++j) {
            aq0[j]     = qrow[kbase + j];
            aq0[8 + j] = qrow[kbase + 8 + j];
            aq1[j]     = qrow[32 + kbase + j];
            aq1[8 + j] = qrow[32 + kbase + 8 + j];
        }
    }

    v8f o0 = {}, o1 = {}, o2 = {}, o3 = {};
    float mrun[8], lrun[8];
    #pragma unroll
    for (int r = 0; r < 8; ++r) { mrun[r] = -__builtin_inff(); lrun[r] = 0.0f; }

    const float SCALE = 0.125f;          // 1/sqrt(64)
    const float LOG2E = 1.44269504088896f;

    // Prologue: stage key tile 0 into buffer 0 (each thread moves 16B of K and V)
    tile_copy16(Kb + (size_t)tid * 8, &Ktile[0][0][0] + tid * 8);
    tile_copy16(Vb + (size_t)tid * 8, &Vtile[0][0][0] + tid * 8);

    for (int kb = 0; kb < NN; kb += 32) {
        const int buf = (kb >> 5) & 1;

        tile_wait();        // own async ops done (no-op on sync fallback)
        __syncthreads();    // everyone's portion of tile[buf] visible

        // Stage next tile into the other buffer while we compute on this one.
        if (kb + 32 < NN) {
            const half_t* ks = Kb + (size_t)(kb + 32) * HDD + tid * 8;
            const half_t* vs = Vb + (size_t)(kb + 32) * HDD + tid * 8;
            tile_copy16(ks, &Ktile[buf ^ 1][0][0] + tid * 8);
            tile_copy16(vs, &Vtile[buf ^ 1][0][0] + tid * 8);
        }

        // S(16x32) = Q(16x64) @ K^T(64x32): two 16-key groups x two 32-d chunks
        v8f s0 = {}, s1 = {};
        s0 = wmma_f16(aq0, *(const v16h*)&Ktile[buf][mrow][hs * 16],          s0);
        s0 = wmma_f16(aq1, *(const v16h*)&Ktile[buf][mrow][32 + hs * 16],     s0);
        s1 = wmma_f16(aq0, *(const v16h*)&Ktile[buf][16 + mrow][hs * 16],     s1);
        s1 = wmma_f16(aq1, *(const v16h*)&Ktile[buf][16 + mrow][32 + hs * 16], s1);

        // Online softmax: row stats via 16-lane butterflies (C layout: reg r = row)
        #pragma unroll
        for (int r = 0; r < 8; ++r) {
            float v0 = s0[r] * SCALE;
            float v1 = s1[r] * SCALE;
            float mx = fmaxf(v0, v1);
            #pragma unroll
            for (int off = 1; off < 16; off <<= 1)
                mx = fmaxf(mx, __shfl_xor(mx, off, 32));
            const float mnew  = fmaxf(mrun[r], mx);
            const float alpha = exp2f((mrun[r] - mnew) * LOG2E);
            const float p0    = exp2f((v0 - mnew) * LOG2E);
            const float p1    = exp2f((v1 - mnew) * LOG2E);
            float rs = p0 + p1;
            #pragma unroll
            for (int off = 1; off < 16; off <<= 1)
                rs += __shfl_xor(rs, off, 32);
            mrun[r] = mnew;
            lrun[r] = lrun[r] * alpha + rs;
            o0[r] *= alpha; o1[r] *= alpha; o2[r] *= alpha; o3[r] *= alpha;
            const int prow = hs * 8 + r;
            Pbuf[wave][prow][mrow]      = (half_t)p0;
            Pbuf[wave][prow][16 + mrow] = (half_t)p1;
        }
        // per-wave LDS RAW: P stores above -> P loads below
        asm volatile("s_wait_dscnt 0" ::: "memory");

        // P(16x32) as A-fragment
        v16h ap;
        {
            const half_t* pr = &Pbuf[wave][mrow][0];
            #pragma unroll
            for (int j = 0; j < 8; ++j) {
                ap[j]     = pr[kbase + j];
                ap[8 + j] = pr[kbase + 8 + j];
            }
        }

        // V B-fragments: B[K=key][N=d] read directly from row-major V tile
        // (b[j] = V[hs*16+j][t*16+mrow]); strided u16 LDS reads, no transpose pass.
        v16h bv0, bv1, bv2, bv3;
        #pragma unroll
        for (int j = 0; j < 16; ++j) {
            const half_t* vr = &Vtile[buf][hs * 16 + j][0];
            bv0[j] = vr[ 0 + mrow];
            bv1[j] = vr[16 + mrow];
            bv2[j] = vr[32 + mrow];
            bv3[j] = vr[48 + mrow];
        }
        o0 = wmma_f16(ap, bv0, o0);
        o1 = wmma_f16(ap, bv1, o1);
        o2 = wmma_f16(ap, bv2, o2);
        o3 = wmma_f16(ap, bv3, o3);
    }

    // Epilogue: out[n][h*64+e] = O / l
    #pragma unroll
    for (int r = 0; r < 8; ++r) {
        const int row = q0 + hs * 8 + r;
        const float inv = 1.0f / lrun[r];
        float* orow = out + (size_t)row * DIMM + h * HDD;
        orow[ 0 + mrow] = o0[r] * inv;
        orow[16 + mrow] = o1[r] * inv;
        orow[32 + mrow] = o2[r] * inv;
        orow[48 + mrow] = o3[r] * inv;
    }
}

extern "C" void kernel_launch(void* const* d_in, const int* in_sizes, int n_in,
                              void* d_out, int out_size, void* d_ws, size_t ws_size,
                              hipStream_t stream) {
    const float* seq = (const float*)d_in[0];
    const float* Wq  = (const float*)d_in[1];
    const float* bq  = (const float*)d_in[2];
    const float* Wk  = (const float*)d_in[3];
    const float* bk  = (const float*)d_in[4];
    const float* Wv  = (const float*)d_in[5];
    const float* bv  = (const float*)d_in[6];
    float* out = (float*)d_out;

    const size_t per = (size_t)HH * NN * HDD;   // 3.1M halves each
    half_t* Qh = (half_t*)d_ws;
    half_t* Kh = Qh + per;
    half_t* Vh = Kh + per;

    dim3 grid(NN / 128, HH);
    qkv_proj_kernel<<<grid, 256, 0, stream>>>(seq, Wq, bq, Wk, bk, Wv, bv, Qh, Kh, Vh);
    attn_fwd_kernel<<<grid, 256, 0, stream>>>(Qh, Kh, Vh, out);
}
